// SelfAttentionLayer_74698071212469
// MI455X (gfx1250) — matrix operand order repaired and checked by reading, not compiled, not run on hardware
//
#include <hip/hip_runtime.h>
#include <hip/hip_bf16.h>

typedef __attribute__((ext_vector_type(16))) _Float16 v16h;
typedef __attribute__((ext_vector_type(8)))  _Float16 v8h;
typedef __attribute__((ext_vector_type(8)))  float    v8f;

// ---------------- workspace layout ----------------
// wqkv fragments: 8 kt * 48 nt * 32 lanes * 16 halves = 196608 halves
// wo   fragments: 8 kt * 16 nt * 32 lanes * 16 halves =  65536 halves
// byte 524288: bqkv[768] f32 ; then bo[256] f32
#define WS_WO_H   196608
#define WS_BIAS_B 524288

// kt strides inside the swizzled fragment arrays (halves)
#define QKV_KT_STRIDE 24576   // 48 ntiles * 512
#define WO_KT_STRIDE  8192    // 16 ntiles * 512

// LDS strides (halves); multiples of 8 keep 16B alignment
#define XS   264   // X/attn, Q, K row stride (256 + 8 pad)
#define VTS  72    // Vt row stride (64 + 8)
#define SS   72    // scores / dist row stride

// LDS map (bytes). Sf (f32 scores, stages 2-3) aliases Xl (x dead after
// stage 1; attn written into Xl only in stage 4, after dist extracted).
// total = 147456 B -> 2 workgroups per 320KB WGP.
#define L_XL   0        // 64 x XS  f16 = 33792 B (x row, later attn; Sf alias)
#define L_QL   33792    // 64 x XS  f16
#define L_KL   67584    // 64 x XS  f16
#define L_VT   101376   // 256 x VTS f16 = 36864 B (V transposed)
#define L_DL   138240   // 64 x SS  f16 = 9216 B  (softmax dist)
#define L_TOT  147456

// ---------------- WMMA fragment helpers (contiguous k order: k = 16*(lane/16) + s) ----------------
static __device__ __forceinline__ v16h ldA(const _Float16* P, int stride, int m0, int k0, int lane) {
  union { v16h v; v8h h[2]; } u;
  const _Float16* p = P + (m0 + (lane & 15)) * stride + k0 + ((lane >> 4) << 4);
  u.h[0] = *(const v8h*)p;
  u.h[1] = *(const v8h*)(p + 8);
  return u.v;
}
// B[k][n] sourced from N-major storage Mat[n][k]
static __device__ __forceinline__ v16h ldBT(const _Float16* P, int stride, int n0, int k0, int lane) {
  union { v16h v; v8h h[2]; } u;
  const _Float16* p = P + (n0 + (lane & 15)) * stride + k0 + ((lane >> 4) << 4);
  u.h[0] = *(const v8h*)p;
  u.h[1] = *(const v8h*)(p + 8);
  return u.v;
}
static __device__ __forceinline__ v8f wmma16(v16h a, v16h b, v8f c) {
  return __builtin_amdgcn_wmma_f32_16x16x32_f16(false, a, false, b, (short)0, c, false, false);
}

// ---------------- prep: fp32 weights -> f16 WMMA-B-swizzled fragments ----------------
__global__ void attn_prep_weights(const float* __restrict__ wq, const float* __restrict__ wk,
                                  const float* __restrict__ wv, const float* __restrict__ wo,
                                  _Float16* __restrict__ ws) {
  int idx  = blockIdx.x * 256 + threadIdx.x;      // 8*64*32*16 = 262144 total
  int t    = idx & 15;
  int lane = (idx >> 4) & 31;
  int nt   = (idx >> 9) & 63;
  int kt   = idx >> 15;
  int k    = kt * 32 + ((lane >> 4) << 4) + t;    // B-frag K index
  if (nt < 48) {                                  // fused wq|wk|wv, N = 0..767
    int n = nt * 16 + (lane & 15);
    const float* w = (n < 256) ? wq : (n < 512) ? wk : wv;
    ws[(size_t)((kt * 48 + nt) * 32 + lane) * 16 + t] = (_Float16)w[k * 256 + (n & 255)];
  } else {                                        // wo
    int n = (nt - 48) * 16 + (lane & 15);
    ws[(size_t)WS_WO_H + (size_t)((kt * 16 + (nt - 48)) * 32 + lane) * 16 + t] =
        (_Float16)wo[k * 256 + n];
  }
}

__global__ void attn_prep_bias(const float* __restrict__ bq, const float* __restrict__ bk,
                               const float* __restrict__ bv, const float* __restrict__ bo,
                               float* __restrict__ wsb) {
  int i = threadIdx.x;                            // 1024 threads
  if (i < 768) wsb[i] = (i < 256) ? bq[i] : (i < 512) ? bk[i - 256] : bv[i - 512];
  else         wsb[i] = bo[i - 768];
}

// ---------------- fused row self-attention: one WG per (b,h) ----------------
__global__ __launch_bounds__(256)
void attn_row_kernel(const float* __restrict__ x,
                     const _Float16* __restrict__ wqkv,
                     const _Float16* __restrict__ wof,
                     const float* __restrict__ bqkv,
                     const float* __restrict__ bof,
                     float* __restrict__ out) {
  extern __shared__ char smem[];
  _Float16* Xl = (_Float16*)(smem + L_XL);
  float*    Sf = (float*)(smem + L_XL);           // alias: lives stages 2-3 only
  _Float16* Ql = (_Float16*)(smem + L_QL);
  _Float16* Kl = (_Float16*)(smem + L_KL);
  _Float16* Vt = (_Float16*)(smem + L_VT);
  _Float16* Dl = (_Float16*)(smem + L_DL);

  const int tid  = threadIdx.x;
  const int wave = tid >> 5;
  const int lane = tid & 31;
  const int mt   = wave & 3;        // this wave's fixed M-tile
  const int half = wave >> 2;       // splits the N range between wave groups
  const size_t g = blockIdx.x;      // (b*64 + h)
  const float* xrow = x + g * 16384;

  // ---- stage 0: load X row, convert to f16 ----
  for (int i = tid; i < 4096; i += 256) {         // 4096 float4s
    float4 f = ((const float4*)xrow)[i];
    int e = i << 2;
    _Float16* d = Xl + (e >> 8) * XS + (e & 255);
    d[0] = (_Float16)f.x; d[1] = (_Float16)f.y;
    d[2] = (_Float16)f.z; d[3] = (_Float16)f.w;
  }
  __syncthreads();

  // ---- stage 1: fused QKV GEMM  [64x256] x [256x768] ----
  // A fragments register-resident; B via one rolling pointer + constant
  // kt byte-offsets (folds into global_load immediate offsets).
  {
    v16h a[8];
    for (int kt = 0; kt < 8; ++kt) a[kt] = ldA(Xl, XS, mt * 16, kt * 32, lane);
    const _Float16* bp = wqkv + ((size_t)(half * 24) * 32 + lane) * 16;
    for (int nt = half * 24; nt < half * 24 + 24; ++nt, bp += 512) {
      v8f acc = {};
#pragma unroll
      for (int kt = 0; kt < 8; ++kt)
        acc = wmma16(a[kt], *(const v16h*)(bp + kt * QKV_KT_STRIDE), acc);
      float bias = bqkv[nt * 16 + (lane & 15)];
      int rb = mt * 16 + ((lane >> 4) << 3);
      int cl = lane & 15;
      if (nt < 16) {                              // Q row-major
        _Float16* dst = Ql + nt * 16 + cl;
        for (int r = 0; r < 8; ++r) dst[(rb + r) * XS] = (_Float16)(acc[r] + bias);
      } else if (nt < 32) {                       // K row-major
        _Float16* dst = Kl + (nt - 16) * 16 + cl;
        for (int r = 0; r < 8; ++r) dst[(rb + r) * XS] = (_Float16)(acc[r] + bias);
      } else {                                    // V transposed: Vt[c][u]
        _Float16* dst = Vt + (size_t)((nt - 32) * 16 + cl) * VTS;
        for (int r = 0; r < 8; ++r) dst[rb + r] = (_Float16)(acc[r] + bias);
      }
    }
  }
  __syncthreads();

  // ---- stage 2: scores = Q . K^T  [64x64 over c=256] -> Sf (aliases Xl) ----
  {
    v16h a[8];
    for (int kt = 0; kt < 8; ++kt) a[kt] = ldA(Ql, XS, mt * 16, kt * 32, lane);
    for (int j = 0; j < 2; ++j) {
      int nt = half * 2 + j;
      v8f acc = {};
      for (int kt = 0; kt < 8; ++kt)
        acc = wmma16(a[kt], ldBT(Kl, XS, nt * 16, kt * 32, lane), acc);
      int rb = mt * 16 + ((lane >> 4) << 3);
      int cl = nt * 16 + (lane & 15);
      for (int r = 0; r < 8; ++r) Sf[(rb + r) * SS + cl] = acc[r];
    }
  }
  __syncthreads();

  // ---- stage 3: stable row softmax (64 rows x 64 cols) ----
  if (tid < 64) {
    float* row = Sf + tid * SS;
    _Float16* dr = Dl + tid * SS;
    float m = row[0];
    for (int u = 1; u < 64; ++u) m = fmaxf(m, row[u]);
    float s = 0.f;
    for (int u = 0; u < 64; ++u) { float e = __expf(row[u] - m); s += e; dr[u] = (_Float16)e; }
    float inv = 1.f / s;
    for (int u = 0; u < 64; ++u) dr[u] = (_Float16)((float)dr[u] * inv);
  }
  __syncthreads();

  // ---- stage 4: attn = dist . V  [64x64]x[64x256] -> Xl (Sf now dead) ----
  {
    v16h a0 = ldA(Dl, SS, mt * 16, 0, lane);
    v16h a1 = ldA(Dl, SS, mt * 16, 32, lane);
    for (int nt = half * 8; nt < half * 8 + 8; ++nt) {
      v8f acc = {};
      acc = wmma16(a0, ldBT(Vt, VTS, nt * 16, 0, lane), acc);
      acc = wmma16(a1, ldBT(Vt, VTS, nt * 16, 32, lane), acc);
      int rb = mt * 16 + ((lane >> 4) << 3);
      _Float16* dst = Xl + nt * 16 + (lane & 15);
      for (int r = 0; r < 8; ++r) dst[(rb + r) * XS] = (_Float16)acc[r];
    }
  }
  __syncthreads();

  // ---- stage 5: out = attn . Wo + bo, fp32 store ----
  {
    v16h a[8];
    for (int kt = 0; kt < 8; ++kt) a[kt] = ldA(Xl, XS, mt * 16, kt * 32, lane);
    float* orow = out + g * 16384;
    const _Float16* bp = wof + ((size_t)(half * 8) * 32 + lane) * 16;
    for (int nt = half * 8; nt < half * 8 + 8; ++nt, bp += 512) {
      v8f acc = {};
#pragma unroll
      for (int kt = 0; kt < 8; ++kt)
        acc = wmma16(a[kt], *(const v16h*)(bp + kt * WO_KT_STRIDE), acc);
      float bias = bof[nt * 16 + (lane & 15)];
      int rb = mt * 16 + ((lane >> 4) << 3);
      int cl = nt * 16 + (lane & 15);
      for (int r = 0; r < 8; ++r) orow[(rb + r) * 256 + cl] = acc[r] + bias;
    }
  }
}

extern "C" void kernel_launch(void* const* d_in, const int* in_sizes, int n_in,
                              void* d_out, int out_size, void* d_ws, size_t ws_size,
                              hipStream_t stream) {
  const float* x  = (const float*)d_in[0];
  const float* wq = (const float*)d_in[1];
  const float* bq = (const float*)d_in[2];
  const float* wk = (const float*)d_in[3];
  const float* bk = (const float*)d_in[4];
  const float* wv = (const float*)d_in[5];
  const float* bv = (const float*)d_in[6];
  const float* wo = (const float*)d_in[7];
  const float* bo = (const float*)d_in[8];
  float* out = (float*)d_out;

  _Float16* wsh = (_Float16*)d_ws;
  float*    wsb = (float*)((char*)d_ws + WS_BIAS_B);

  attn_prep_weights<<<1024, 256, 0, stream>>>(wq, wk, wv, wo, wsh);
  attn_prep_bias<<<1, 1024, 0, stream>>>(bq, bk, bv, bo, wsb);

  attn_row_kernel<<<2048, 256, L_TOT, stream>>>(
      x, wsh, wsh + WS_WO_H, wsb, wsb + 768, out);
}
